// PatchAdapterLayer_18442589569380
// MI455X (gfx1250) — compile-verified, matrix operand order checked by reading, not run on hardware
//
#include <hip/hip_runtime.h>
#include <hip/hip_bf16.h>
#include <math.h>

// ---------------------------------------------------------------------------
// PatchAdapterLayer for MI455X (gfx1250): router + 4 FFT-attention experts.
// All 1x1 convs run as v_wmma_f32_16x16x32_bf16 GEMMs from LDS; depthwise and
// circular-conv stages are f32 VALU on LDS tiles. One workgroup (8 wave32)
// per 16x16 patch; 288KB dynamic LDS (WGP 320KB budget).
//   - A-fragment reuse: each wave owns one M-tile, streams 8 N-tiles.
//   - Final GEMM fuses residual + gate + global store/accumulate.
// ---------------------------------------------------------------------------

typedef __attribute__((ext_vector_type(16))) __bf16        v16bf;
typedef __attribute__((ext_vector_type(8)))  float         v8f;
typedef __attribute__((ext_vector_type(4)))  unsigned int  v4u;

static_assert(sizeof(v16bf) == 32, "v16bf must be 32 bytes");

#define NPATCH   2048      // B(8) * P(256)
#define DIMC     96
#define RANKR    64
#define SMEM_XB      0                         // bf16 [256][96]   48KB
#define SMEM_HB      49152                     // bf16 [256][64]   32KB
#define SMEM_F0      81920                     // f32  [64][256]   64KB
#define SMEM_F1      147456
#define SMEM_F2      212992
#define SMEM_WB      278528                    // staging          16KB
#define SMEM_BYTES   294912

// ---- WMMA fragment load: operand stored [row][k] row-major in LDS (bf16) ---
// A-operand: row = M (lane%16), K halves split by lane/16.
// B-operand: row = N (lane%16), same K mapping (pixel-major activation store).
__device__ inline v16bf load_frag(const __bf16* base, int ldk, int row, int k0,
                                  int lane) {
  union { v16bf v; v4u q[2]; } f;
  const __bf16* p = base + (size_t)row * ldk + k0 + ((lane >> 4) << 3);
  f.q[0] = *(const v4u*)(p);        // K = k0 + h*8 .. +7       (ds_load_b128)
  f.q[1] = *(const v4u*)(p + 16);   // K = k0 + 16 + h*8 .. +7  (ds_load_b128)
  return f.v;
}

__device__ inline void stage_w(__bf16* wb, const float* g, int count, int tid) {
  for (int t = tid; t < count; t += 256) wb[t] = (__bf16)g[t];
}
__device__ inline void stage_wf(float* wf, const float* g, int count, int tid) {
  for (int t = tid; t < count; t += 256) wf[t] = g[t];
}

// GEMM: D[64 x 256] = A[64 x K] * B[K x 256]; D written f32 channel-major.
// Wave w owns M-tile (w&3) and N-tiles (w>>2)*8 .. +7: A-fragment loaded once
// per k-step and reused for 8 WMMAs.
template <int K>
__device__ void wmma_gemm_f32(const __bf16* A, const __bf16* B, float* D,
                              const float* bias, int tid) {
  const int wid = tid >> 5, lane = tid & 31;
  const int mt = wid & 3, g = wid >> 2;
  v8f acc[8] = {};
#pragma unroll
  for (int kk = 0; kk < K; kk += 32) {
    const v16bf a = load_frag(A, K, mt * 16 + (lane & 15), kk, lane);
#pragma unroll
    for (int i = 0; i < 8; ++i) {
      const v16bf b =
          load_frag(B, K, (g * 8 + i) * 16 + (lane & 15), kk, lane);
      acc[i] = __builtin_amdgcn_wmma_f32_16x16x32_bf16(
          false, a, false, b, (short)0, acc[i], false, false);
    }
  }
  const int m0 = mt * 16 + ((lane >> 4) << 3);
#pragma unroll
  for (int i = 0; i < 8; ++i) {
    const int n = (g * 8 + i) * 16 + (lane & 15);
#pragma unroll
    for (int r = 0; r < 8; ++r) {
      float v = acc[i][r];
      if (bias) v += bias[m0 + r];
      D[(m0 + r) * 256 + n] = v;
    }
  }
}

// Same GEMM with bf16 output, pixel-major [256][64] (feeds next GEMM's B).
template <int K>
__device__ void wmma_gemm_bf(const __bf16* A, const __bf16* B, __bf16* Db,
                             int tid) {
  const int wid = tid >> 5, lane = tid & 31;
  const int mt = wid & 3, g = wid >> 2;
  v8f acc[8] = {};
#pragma unroll
  for (int kk = 0; kk < K; kk += 32) {
    const v16bf a = load_frag(A, K, mt * 16 + (lane & 15), kk, lane);
#pragma unroll
    for (int i = 0; i < 8; ++i) {
      const v16bf b =
          load_frag(B, K, (g * 8 + i) * 16 + (lane & 15), kk, lane);
      acc[i] = __builtin_amdgcn_wmma_f32_16x16x32_bf16(
          false, a, false, b, (short)0, acc[i], false, false);
    }
  }
  const int m0 = mt * 16 + ((lane >> 4) << 3);
#pragma unroll
  for (int i = 0; i < 8; ++i) {
    const int n = (g * 8 + i) * 16 + (lane & 15);
#pragma unroll
    for (int r = 0; r < 8; ++r) Db[n * RANKR + m0 + r] = (__bf16)acc[i][r];
  }
}

// Final GEMM [96x64]x[64x256]: out = gate * (p2 @ h2 + shortcut) fused into
// global store/accumulate. Wave owns 2 N-columns; B-fragment reused over the
// 6 M-tiles of each column.
template <bool FIRST>
__device__ void wmma_gemm_out(const __bf16* A /*[96][64]*/, const __bf16* B,
                              const float* xg, float* outg, float gate, int b,
                              int ph, int pc, int tid) {
  const int wid = tid >> 5, lane = tid & 31;
#pragma unroll
  for (int half = 0; half < 2; ++half) {
    const int nt = wid * 2 + half;
    v8f acc[6] = {};
#pragma unroll
    for (int kk = 0; kk < 64; kk += 32) {
      const v16bf bfr = load_frag(B, 64, nt * 16 + (lane & 15), kk, lane);
#pragma unroll
      for (int mt = 0; mt < 6; ++mt) {
        const v16bf a = load_frag(A, 64, mt * 16 + (lane & 15), kk, lane);
        acc[mt] = __builtin_amdgcn_wmma_f32_16x16x32_bf16(
            false, a, false, bfr, (short)0, acc[mt], false, false);
      }
    }
    const int n = nt * 16 + (lane & 15);
    const int y = n >> 4, xx = n & 15;
#pragma unroll
    for (int mt = 0; mt < 6; ++mt) {
      const int m0 = mt * 16 + ((lane >> 4) << 3);
#pragma unroll
      for (int r = 0; r < 8; ++r) {
        const int m = m0 + r;
        const size_t gi =
            (((size_t)b * DIMC + m) * 256 + ph * 16 + y) * 256 + pc * 16 + xx;
        const float val = gate * (acc[mt][r] + xg[gi]);
        if (FIRST) outg[gi] = val; else outg[gi] += val;
      }
    }
  }
}

// Depthwise KSxKS conv with SAME zero padding inside the 16x16 patch.
template <int KS>
__device__ void dwconv(const float* src, float* dst, const float* wlds,
                       const float* biasg, int tid) {
  const int y = tid >> 4, xx = tid & 15;
  for (int c = 0; c < RANKR; ++c) {
    float acc = biasg[c];
#pragma unroll
    for (int dy = 0; dy < KS; ++dy) {
      const int yy = y + dy - KS / 2;
      if (yy < 0 || yy >= 16) continue;
#pragma unroll
      for (int dx = 0; dx < KS; ++dx) {
        const int xc = xx + dx - KS / 2;
        if (xc < 0 || xc >= 16) continue;
        acc += src[c * 256 + yy * 16 + xc] * wlds[c * KS * KS + dy * KS + dx];
      }
    }
    dst[c * 256 + tid] = acc;
  }
}

// Per-channel PxP circular convolution (== irfft2(rfft2(q)*rfft2(k))).
// For P=32 the patch is zero-padded, so terms with k-index >= 16 vanish.
template <int P>
__device__ void circconv(const float* q, const float* k, float* dst, int tid) {
  constexpr int I = (P < 16) ? P : 16;
  const int y = tid >> 4, xx = tid & 15;
  const int m = y & (P - 1), nn = xx & (P - 1);
  const int y0 = y - m, x0 = xx - nn;
  for (int c = 0; c < RANKR; ++c) {
    float acc = 0.0f;
    for (int i = 0; i < I; ++i) {
      const int ky = (m - i) & (P - 1);
      if (P > 16 && ky >= 16) continue;
      for (int j = 0; j < I; ++j) {
        const int kx = (nn - j) & (P - 1);
        if (P > 16 && kx >= 16) continue;
        acc += q[c * 256 + (y0 + i) * 16 + (x0 + j)] *
               k[c * 256 + (y0 + ky) * 16 + (x0 + kx)];
      }
    }
    dst[c * 256 + tid] = acc;
  }
}

// ---------------------------------------------------------------------------
// Router: per-patch logits (patch . gate_w[e]) + freq term, softmax, top-2.
// ---------------------------------------------------------------------------
__global__ __launch_bounds__(256) void router_kernel(
    const float* __restrict__ x, const float* __restrict__ gate_w,
    const float* __restrict__ gate_b, const float* __restrict__ freq_emb,
    const float* __restrict__ freq_w, float* __restrict__ gates) {
  __shared__ float red[256 * 4];
  const int tid = threadIdx.x;
  const int n = blockIdx.x;
  const int b = n >> 8, pi = n & 255, ph = pi >> 4, pc = pi & 15;

  float acc[4] = {0.f, 0.f, 0.f, 0.f};
  for (int t = tid; t < DIMC * 256; t += 256) {
    const int c = t >> 8, r = (t >> 4) & 15, s = t & 15;
    const float xv =
        x[(((size_t)b * DIMC + c) * 256 + ph * 16 + r) * 256 + pc * 16 + s];
#pragma unroll
    for (int e = 0; e < 4; ++e)
      acc[e] += xv * gate_w[((e * DIMC + c) * 16 + r) * 16 + s];
  }
#pragma unroll
  for (int e = 0; e < 4; ++e) red[tid * 4 + e] = acc[e];
  __syncthreads();
  for (int s = 128; s > 0; s >>= 1) {
    if (tid < s)
#pragma unroll
      for (int e = 0; e < 4; ++e) red[tid * 4 + e] += red[(tid + s) * 4 + e];
    __syncthreads();
  }
  if (tid == 0) {
    float lg[4];
#pragma unroll
    for (int e = 0; e < 4; ++e) {
      float ft = 0.f;
      for (int f = 0; f < 128; ++f)
        ft += freq_emb[b * 128 + f] * freq_w[e * 128 + f];
      lg[e] = red[e] + gate_b[e] + ft;
    }
    const float mx = fmaxf(fmaxf(lg[0], lg[1]), fmaxf(lg[2], lg[3]));
    float ex[4], ssum = 0.f;
#pragma unroll
    for (int e = 0; e < 4; ++e) { ex[e] = __expf(lg[e] - mx); ssum += ex[e]; }
#pragma unroll
    for (int e = 0; e < 4; ++e) ex[e] /= ssum;
    int i1 = 0;
    for (int e = 1; e < 4; ++e) if (ex[e] > ex[i1]) i1 = e;
    int i2 = -1;
    for (int e = 0; e < 4; ++e)
      if (e != i1 && (i2 < 0 || ex[e] > ex[i2])) i2 = e;
#pragma unroll
    for (int e = 0; e < 4; ++e)
      gates[n * 4 + e] = (e == i1) ? ex[i1] : (e == i2) ? ex[i2] : 0.f;
  }
}

// ---------------------------------------------------------------------------
// Expert kernel: one workgroup per patch. P = FFT block size. FIRST = store
// (expert 0 initializes d_out) vs accumulate.
// ---------------------------------------------------------------------------
template <int P, bool FIRST>
__global__ __launch_bounds__(256, 1) void expert_kernel(
    const float* __restrict__ x, const float* __restrict__ gates,
    float* __restrict__ outg, int e,
    const float* __restrict__ p0, const float* __restrict__ p1,
    const float* __restrict__ p2, const float* __restrict__ qw,
    const float* __restrict__ qdw, const float* __restrict__ qdb,
    const float* __restrict__ kvw, const float* __restrict__ kvdw,
    const float* __restrict__ kvdb, const float* __restrict__ lnw,
    const float* __restrict__ lnb, const float* __restrict__ pw,
    const float* __restrict__ pb) {
  extern __shared__ char smem[];
  __bf16* xb = (__bf16*)(smem + SMEM_XB);   // [256][96]  pixel-major
  __bf16* hb = (__bf16*)(smem + SMEM_HB);   // [256][64]  pixel-major
  float*  f0 = (float*)(smem + SMEM_F0);    // [64][256]  channel-major
  float*  f1 = (float*)(smem + SMEM_F1);
  float*  f2 = (float*)(smem + SMEM_F2);
  __bf16* wb = (__bf16*)(smem + SMEM_WB);   // weight staging (bf16 view)
  float*  wf = (float*)(smem + SMEM_WB);    // weight staging (f32 view)

  const int tid = threadIdx.x;
  const int n = blockIdx.x;
  const int b = n >> 8, pi = n & 255, ph = pi >> 4, pc = pi & 15;
  const float gate = gates[n * 4 + e];

  if (gate == 0.0f) {           // top-2 sparsity: zero contribution
    if (FIRST) {
      // coalesced: adjacent lanes write adjacent pixels
      for (int t = tid; t < DIMC * 256; t += 256) {
        const int c = t >> 8, pix = t & 255, y = pix >> 4, xx = pix & 15;
        outg[(((size_t)b * DIMC + c) * 256 + ph * 16 + y) * 256 + pc * 16 + xx]
            = 0.0f;
      }
    }
    return;
  }

  // Warm caches for this expert's weights (global_prefetch_b8).
  {
    const int cl = tid * 16;  // one 64B line per thread across each tensor
    if (cl < 2 * RANKR * RANKR) __builtin_prefetch(kvw + cl, 0, 3);
    if (cl < RANKR * DIMC) {
      __builtin_prefetch(p0 + cl, 0, 3);
      __builtin_prefetch(p1 + cl, 0, 3);
      __builtin_prefetch(p2 + cl, 0, 3);
    }
    if (cl < RANKR * RANKR) {
      __builtin_prefetch(qw + cl, 0, 3);
      __builtin_prefetch(pw + cl, 0, 3);
    }
    if (cl < 2 * RANKR * 49) __builtin_prefetch(kvdw + cl, 0, 3);
  }

  // Stage x patch -> LDS bf16, pixel-major [256][96]. Adjacent lanes read
  // adjacent pixels (contiguous 64B rows of x) for coalescing; the scattered
  // side lands in LDS (ds_store_b16), which is cheap.
  for (int t = tid; t < DIMC * 256; t += 256) {
    const int c = t >> 8, pix = t & 255, y = pix >> 4, xx = pix & 15;
    xb[pix * DIMC + c] = (__bf16)
        x[(((size_t)b * DIMC + c) * 256 + ph * 16 + y) * 256 + pc * 16 + xx];
  }
  __syncthreads();

  // G1: h = p0 @ x  -> hb (bf16)
  stage_w(wb, p0, RANKR * DIMC, tid); __syncthreads();
  wmma_gemm_bf<DIMC>(wb, xb, hb, tid); __syncthreads();

  // G2: qpre = qw @ h -> f0 ; dw3(qpre)+qdb -> f1 (q)
  stage_w(wb, qw, RANKR * RANKR, tid); __syncthreads();
  wmma_gemm_f32<RANKR>(wb, hb, f0, nullptr, tid); __syncthreads();
  stage_wf(wf, qdw, RANKR * 9, tid); __syncthreads();
  dwconv<3>(f0, f1, wf, qdb, tid); __syncthreads();

  // G3a: kpre = kvw[0:64] @ h -> f0 ; dw7+bias -> f2 (k)
  stage_w(wb, kvw, RANKR * RANKR, tid); __syncthreads();
  wmma_gemm_f32<RANKR>(wb, hb, f0, nullptr, tid); __syncthreads();
  stage_wf(wf, kvdw, RANKR * 49, tid); __syncthreads();
  dwconv<7>(f0, f2, wf, kvdb, tid); __syncthreads();

  // FFT-attention core: per-channel PxP circular conv of q,k -> f0
  circconv<P>(f1, f2, f0, tid); __syncthreads();

  // G3b: vpre = kvw[64:128] @ h -> f1 ; dw7+bias -> f2 (v)
  stage_w(wb, kvw + RANKR * RANKR, RANKR * RANKR, tid); __syncthreads();
  wmma_gemm_f32<RANKR>(wb, hb, f1, nullptr, tid); __syncthreads();
  stage_wf(wf, kvdw + RANKR * 49, RANKR * 49, tid); __syncthreads();
  dwconv<7>(f1, f2, wf, kvdb + RANKR, tid); __syncthreads();

  // LayerNorm over channels, * v -> hb (bf16, pixel-major)
  {
    float s = 0.f, s2 = 0.f;
    for (int c = 0; c < RANKR; ++c) {
      const float t = f0[c * 256 + tid];
      s += t; s2 += t * t;
    }
    const float mu = s * (1.0f / RANKR);
    const float var = s2 * (1.0f / RANKR) - mu * mu;
    const float inv = rsqrtf(var + 1e-5f);
    for (int c = 0; c < RANKR; ++c) {
      const float t = (f0[c * 256 + tid] - mu) * inv * lnw[c] + lnb[c];
      hb[tid * RANKR + c] = (__bf16)(t * f2[c * 256 + tid]);
    }
  }
  __syncthreads();

  // G4: attn = pw @ normed + pb -> f0
  stage_w(wb, pw, RANKR * RANKR, tid); __syncthreads();
  wmma_gemm_f32<RANKR>(wb, hb, f0, pb, tid); __syncthreads();

  // G5: g = p1 @ x -> f1 ; h2 = attn * silu(g) -> hb
  stage_w(wb, p1, RANKR * DIMC, tid); __syncthreads();
  wmma_gemm_f32<DIMC>(wb, xb, f1, nullptr, tid); __syncthreads();
  for (int c = 0; c < RANKR; ++c) {
    const float a = f0[c * 256 + tid];
    const float g = f1[c * 256 + tid];
    hb[tid * RANKR + c] = (__bf16)(a * (g / (1.0f + __expf(-g))));
  }
  __syncthreads();

  // G6: out = gate * (p2 @ h2 + x)  fused store/accumulate to global.
  stage_w(wb, p2, DIMC * RANKR, tid); __syncthreads();
  wmma_gemm_out<FIRST>(wb, hb, x, outg, gate, b, ph, pc, tid);
}

// ---------------------------------------------------------------------------
extern "C" void kernel_launch(void* const* d_in, const int* in_sizes, int n_in,
                              void* d_out, int out_size, void* d_ws,
                              size_t ws_size, hipStream_t stream) {
  (void)in_sizes; (void)n_in; (void)out_size; (void)ws_size;
  const float* x        = (const float*)d_in[0];
  const float* freq_emb = (const float*)d_in[1];
  const float* gate_w   = (const float*)d_in[2];
  const float* gate_b   = (const float*)d_in[3];
  const float* freq_w   = (const float*)d_in[4];
  const float* proj0    = (const float*)d_in[5];
  const float* proj1    = (const float*)d_in[6];
  const float* proj2    = (const float*)d_in[7];
  const float* qw       = (const float*)d_in[8];
  const float* qdw      = (const float*)d_in[9];
  const float* qdb      = (const float*)d_in[10];
  const float* kvw      = (const float*)d_in[11];
  const float* kvdw     = (const float*)d_in[12];
  const float* kvdb     = (const float*)d_in[13];
  const float* lnw      = (const float*)d_in[14];
  const float* lnb      = (const float*)d_in[15];
  const float* pow_     = (const float*)d_in[16];
  const float* pob      = (const float*)d_in[17];
  float* out = (float*)d_out;
  float* gates = (float*)d_ws;   // [2048][4]

  router_kernel<<<NPATCH, 256, 0, stream>>>(x, gate_w, gate_b, freq_emb,
                                            freq_w, gates);

  hipFuncSetAttribute(reinterpret_cast<const void*>(&expert_kernel<4, true>),
                      hipFuncAttributeMaxDynamicSharedMemorySize, SMEM_BYTES);
  hipFuncSetAttribute(reinterpret_cast<const void*>(&expert_kernel<8, false>),
                      hipFuncAttributeMaxDynamicSharedMemorySize, SMEM_BYTES);
  hipFuncSetAttribute(reinterpret_cast<const void*>(&expert_kernel<16, false>),
                      hipFuncAttributeMaxDynamicSharedMemorySize, SMEM_BYTES);
  hipFuncSetAttribute(reinterpret_cast<const void*>(&expert_kernel<32, false>),
                      hipFuncAttributeMaxDynamicSharedMemorySize, SMEM_BYTES);

#define EXPERT_ARGS(E)                                                        \
  x, gates, out, E, proj0 + (E)*RANKR * DIMC, proj1 + (E)*RANKR * DIMC,       \
      proj2 + (E)*DIMC * RANKR, qw + (E)*RANKR * RANKR, qdw + (E)*RANKR * 9,  \
      qdb + (E)*RANKR, kvw + (E)*2 * RANKR * RANKR, kvdw + (E)*2 * RANKR * 49,\
      kvdb + (E)*2 * RANKR, lnw + (E)*RANKR, lnb + (E)*RANKR,                 \
      pow_ + (E)*RANKR * RANKR, pob + (E)*RANKR

  expert_kernel<4, true><<<NPATCH, 256, SMEM_BYTES, stream>>>(EXPERT_ARGS(0));
  expert_kernel<8, false><<<NPATCH, 256, SMEM_BYTES, stream>>>(EXPERT_ARGS(1));
  expert_kernel<16, false><<<NPATCH, 256, SMEM_BYTES, stream>>>(EXPERT_ARGS(2));
  expert_kernel<32, false><<<NPATCH, 256, SMEM_BYTES, stream>>>(EXPERT_ARGS(3));
#undef EXPERT_ARGS
}
